// BahdanauAttention_63539746177178
// MI455X (gfx1250) — matrix operand order repaired and checked by reading, not compiled, not run on hardware
//
#include <hip/hip_runtime.h>

#define B_  32
#define S_  2048
#define QH_ 1024
#define KH_ 1024
#define H_  512

typedef __attribute__((ext_vector_type(16))) __bf16        v16bf;
typedef __attribute__((ext_vector_type(8)))  float         v8f;
typedef __attribute__((ext_vector_type(4)))  unsigned int  u32x4;

union FragU { u32x4 u[2]; v16bf bf; };

__device__ __forceinline__ unsigned short f2bf(float f) {
  unsigned u = __float_as_uint(f);
  u += 0x7fffu + ((u >> 16) & 1u);   // round-to-nearest-even
  return (unsigned short)(u >> 16);
}

// ---------------------------------------------------------------------------
// Kernel 1a: q_proj[b][h] = query[b] . Wa_w[h] + Wa_b[h] + Ua_b[h]
// (Ua_b folded in since it is summed before tanh anyway)
// ---------------------------------------------------------------------------
__global__ __launch_bounds__(512) void qproj_kernel(
    const float* __restrict__ query, const float* __restrict__ Wa_w,
    const float* __restrict__ Wa_b, const float* __restrict__ Ua_b,
    float* __restrict__ qp)
{
  __shared__ float q[QH_];
  const int b = blockIdx.x, tid = threadIdx.x;
  for (int d = tid; d < QH_; d += 512) q[d] = query[b * QH_ + d];
  __syncthreads();
  const float4* wr = (const float4*)(Wa_w + (size_t)tid * QH_);
  float acc = 0.f;
  for (int d4 = 0; d4 < QH_ / 4; ++d4) {
    float4 w = wr[d4];
    acc += w.x * q[d4*4+0] + w.y * q[d4*4+1] + w.z * q[d4*4+2] + w.w * q[d4*4+3];
  }
  qp[b * H_ + tid] = acc + Wa_b[tid] + Ua_b[tid];
}

// ---------------------------------------------------------------------------
// Kernel 1b: Ua_w fp32 -> bf16 (row-major [H][KH]), done once, L2-resident
// ---------------------------------------------------------------------------
__global__ __launch_bounds__(256) void cvt_bf16_kernel(
    const float* __restrict__ src, unsigned short* __restrict__ dst, int n4)
{
  int i = blockIdx.x * blockDim.x + threadIdx.x;
  if (i < n4) {
    float4 v = ((const float4*)src)[i];
    ushort4 o;
    o.x = f2bf(v.x); o.y = f2bf(v.y); o.z = f2bf(v.z); o.w = f2bf(v.w);
    ((ushort4*)dst)[i] = o;
  }
}

// ---------------------------------------------------------------------------
// Kernel 2: fused  scores[b][s] = sum_h Va[h]*tanh(keys[b,s].Ua[h] + qp[b][h])
// WG = 512 threads (16 waves), owns 64 s-rows x all 512 h.
// keys tile staged fp32->bf16 in LDS; bf16 WMMA 16x16x32, f32 accum.
// wave w: m-tile = w>>2 (16 rows), column group = w&3 (8 n-tiles = 128 h).
// ---------------------------------------------------------------------------
__global__ __launch_bounds__(512, 1) void scores_kernel(
    const float* __restrict__ keys, const unsigned short* __restrict__ UaB,
    const float* __restrict__ qp, const float* __restrict__ Va_w,
    const float* __restrict__ Va_b, float* __restrict__ scores)
{
  extern __shared__ char smem[];
  unsigned short* ldsK   = (unsigned short*)smem;            // 64*1024 bf16 = 128KB
  float*          ldsPart = (float*)(smem + 64 * 1024 * 2);  // 64*4 floats

  const int tid  = threadIdx.x;
  const int lane = tid & 31;
  const int wave = tid >> 5;
  const int b    = blockIdx.x >> 5;
  const int s0   = (blockIdx.x & 31) * 64;

  // stage keys rows s0..s0+63 (fp32 -> bf16) into LDS, coalesced float4
  const float4* kbase = (const float4*)(keys + ((size_t)b * S_ + s0) * (size_t)KH_);
  for (int i = tid; i < 64 * KH_ / 4; i += 512) {
    float4 v = kbase[i];
    ushort4 o;
    o.x = f2bf(v.x); o.y = f2bf(v.y); o.z = f2bf(v.z); o.w = f2bf(v.w);
    *(ushort4*)(ldsK + i * 4) = o;
  }
  __syncthreads();

  const int mt  = wave >> 2;
  const int col = wave & 3;
  const int m0  = mt * 16;
  const int n   = lane & 15;
  const int hiG = lane >> 4;           // half-wave group (0/1)

  // A fragment: row m0+n of LDS keys; K chunks at +kb and +kb+16 (ISA layout)
  const unsigned short* Arow = ldsK + (m0 + n) * KH_;
  const int kb = hiG * 8;
  // B fragment: column h = col*128 + j*16 + n of Ua (row h contiguous in K);
  // lane group selects K sub-chunk base hiG*8 (chunks at +0 and +16)
  const unsigned short* Bbase = UaB + (size_t)(col * 128 + n) * KH_ + hiG * 8;

  v8f c[8];
  #pragma unroll
  for (int j = 0; j < 8; ++j) {
    v8f z = {0.f, 0.f, 0.f, 0.f, 0.f, 0.f, 0.f, 0.f};
    c[j] = z;
  }

  for (int k0 = 0; k0 < KH_; k0 += 32) {
    FragU fa;
    fa.u[0] = *(const u32x4*)(Arow + k0 + kb);        // K = kb .. kb+7
    fa.u[1] = *(const u32x4*)(Arow + k0 + kb + 16);   // K = kb+16 .. kb+23
    #pragma unroll
    for (int j = 0; j < 8; ++j) {
      const unsigned short* bp = Bbase + (size_t)j * 16 * KH_ + k0;
      FragU fb;
      fb.u[0] = *(const u32x4*)(bp);
      fb.u[1] = *(const u32x4*)(bp + 16);
      c[j] = __builtin_amdgcn_wmma_f32_16x16x32_bf16(
                 false, fa.bf, false, fb.bf, (short)0, c[j], false, false);
    }
  }

  // epilogue: partial score per row = sum over this wave's 128 h-columns
  float sacc[8];
  #pragma unroll
  for (int r = 0; r < 8; ++r) sacc[r] = 0.f;

  const float* qpB = qp   + b * H_ + col * 128 + n;
  const float* VaP = Va_w + col * 128 + n;
  #pragma unroll
  for (int j = 0; j < 8; ++j) {
    float qv = qpB[j * 16];
    float vw = VaP[j * 16];
    #pragma unroll
    for (int r = 0; r < 8; ++r)
      sacc[r] += vw * tanhf(c[j][r] + qv);   // c[j][r] -> row m0 + r + 8*hiG
  }
  // reduce across the 16 lanes (n) inside each half-wave group
  #pragma unroll
  for (int r = 0; r < 8; ++r) {
    float v = sacc[r];
    v += __shfl_xor(v, 1, 32);
    v += __shfl_xor(v, 2, 32);
    v += __shfl_xor(v, 4, 32);
    v += __shfl_xor(v, 8, 32);
    sacc[r] = v;
  }
  if (n == 0) {                          // lanes 0 and 16 hold group sums
    int mbase = m0 + hiG * 8;
    #pragma unroll
    for (int r = 0; r < 8; ++r)
      ldsPart[(mbase + r) * 4 + col] = sacc[r];
  }
  __syncthreads();
  if (tid < 64) {
    float s = ldsPart[tid*4+0] + ldsPart[tid*4+1] +
              ldsPart[tid*4+2] + ldsPart[tid*4+3] + Va_b[0];
    scores[b * S_ + s0 + tid] = s;
  }
}

// ---------------------------------------------------------------------------
// Kernel 3: softmax over S per batch -> weights (written to d_out region)
// ---------------------------------------------------------------------------
__global__ __launch_bounds__(256) void softmax_kernel(
    const float* __restrict__ scores, float* __restrict__ wout)
{
  __shared__ float red[256];
  const int b = blockIdx.x, tid = threadIdx.x;
  const float* sb = scores + b * S_;
  float vals[8];
  float m = -3.402823466e38f;
  #pragma unroll
  for (int i = 0; i < 8; ++i) { vals[i] = sb[tid + i * 256]; m = fmaxf(m, vals[i]); }
  red[tid] = m; __syncthreads();
  for (int off = 128; off > 0; off >>= 1) {
    if (tid < off) red[tid] = fmaxf(red[tid], red[tid + off]);
    __syncthreads();
  }
  float mx = red[0]; __syncthreads();
  float sum = 0.f;
  #pragma unroll
  for (int i = 0; i < 8; ++i) { vals[i] = expf(vals[i] - mx); sum += vals[i]; }
  red[tid] = sum; __syncthreads();
  for (int off = 128; off > 0; off >>= 1) {
    if (tid < off) red[tid] += red[tid + off];
    __syncthreads();
  }
  float inv = 1.f / red[0];
  #pragma unroll
  for (int i = 0; i < 8; ++i) wout[b * S_ + tid + i * 256] = vals[i] * inv;
}

// ---------------------------------------------------------------------------
// Kernel 4: partial context: part[b][sr][d] = sum_{s in range} w[b,s]*keys[b,s,d]
// Second (and last) full pass over keys; fully coalesced float4.
// ---------------------------------------------------------------------------
__global__ __launch_bounds__(256) void ctx_partial_kernel(
    const float* __restrict__ keys, const float* __restrict__ w,
    float* __restrict__ part)
{
  const int b = blockIdx.x >> 3, sr = blockIdx.x & 7, tid = threadIdx.x;
  const float* kb = keys + ((size_t)b * S_ + sr * 256) * (size_t)KH_;
  const float* wb = w + b * S_ + sr * 256;
  float4 acc = {0.f, 0.f, 0.f, 0.f};
  for (int s = 0; s < 256; ++s) {
    float wv = wb[s];
    float4 kv = *(const float4*)(kb + (size_t)s * KH_ + tid * 4);
    acc.x += wv * kv.x; acc.y += wv * kv.y; acc.z += wv * kv.z; acc.w += wv * kv.w;
  }
  *(float4*)(part + (size_t)blockIdx.x * KH_ + tid * 4) = acc;
}

// ---------------------------------------------------------------------------
// Kernel 5: context[b][h] = (sum_p part[b][p]) . Ya_w[h] + Ya_b[h]
// ---------------------------------------------------------------------------
__global__ __launch_bounds__(512) void ya_proj_kernel(
    const float* __restrict__ part, const float* __restrict__ Ya_w,
    const float* __restrict__ Ya_b, float* __restrict__ outctx)
{
  __shared__ float ctx[KH_];
  const int b = blockIdx.x, tid = threadIdx.x;
  for (int d = tid; d < KH_; d += 512) {
    float s = 0.f;
    #pragma unroll
    for (int p = 0; p < 8; ++p) s += part[(size_t)(b * 8 + p) * KH_ + d];
    ctx[d] = s;
  }
  __syncthreads();
  const float4* wr = (const float4*)(Ya_w + (size_t)tid * KH_);
  float acc = 0.f;
  for (int d4 = 0; d4 < KH_ / 4; ++d4) {
    float4 w = wr[d4];
    acc += w.x * ctx[d4*4+0] + w.y * ctx[d4*4+1] +
           w.z * ctx[d4*4+2] + w.w * ctx[d4*4+3];
  }
  outctx[b * H_ + tid] = acc + Ya_b[tid];
}

// ---------------------------------------------------------------------------
extern "C" void kernel_launch(void* const* d_in, const int* in_sizes, int n_in,
                              void* d_out, int out_size, void* d_ws, size_t ws_size,
                              hipStream_t stream) {
  const float* query = (const float*)d_in[0];
  const float* keys  = (const float*)d_in[1];
  const float* Wa_w  = (const float*)d_in[2];
  const float* Wa_b  = (const float*)d_in[3];
  const float* Ua_w  = (const float*)d_in[4];
  const float* Ua_b  = (const float*)d_in[5];
  const float* Va_w  = (const float*)d_in[6];
  const float* Va_b  = (const float*)d_in[7];
  const float* Ya_w  = (const float*)d_in[8];
  const float* Ya_b  = (const float*)d_in[9];

  char* ws = (char*)d_ws;
  float*          qp     = (float*)(ws);                                   // 64KB
  float*          scores = (float*)(ws + (64 << 10));                      // 256KB
  float*          part   = (float*)(ws + (64 << 10) + (256 << 10));        // 1MB
  unsigned short* UaB    = (unsigned short*)(ws + (64 << 10) + (256 << 10) + (1 << 20)); // 1MB

  float* out_ctx = (float*)d_out;            // [32,512]
  float* out_w   = (float*)d_out + B_ * H_;  // [32,2048]

  qproj_kernel<<<B_, 512, 0, stream>>>(query, Wa_w, Wa_b, Ua_b, qp);
  cvt_bf16_kernel<<<(H_ * KH_ / 4 + 255) / 256, 256, 0, stream>>>(Ua_w, UaB, H_ * KH_ / 4);

  const size_t smem2 = 64 * KH_ * 2 + 64 * 4 * sizeof(float);  // 129KB dynamic LDS
  scores_kernel<<<B_ * (S_ / 64), 512, smem2, stream>>>(keys, UaB, qp, Va_w, Va_b, scores);

  softmax_kernel<<<B_, 256, 0, stream>>>(scores, out_w);
  ctx_partial_kernel<<<B_ * 8, 256, 0, stream>>>(keys, out_w, part);
  ya_proj_kernel<<<B_, 512, 0, stream>>>(part, Ya_w, Ya_b, out_ctx);
}
